// LSTM_23596550324313
// MI455X (gfx1250) — matrix operand order repaired and checked by reading, not compiled, not run on hardware
//
#include <hip/hip_runtime.h>
#include <hip/hip_bf16.h>

typedef __attribute__((ext_vector_type(16))) __bf16 v16bf;
typedef __attribute__((ext_vector_type(8)))  __bf16 v8bf;
typedef __attribute__((ext_vector_type(8)))  float  v8f;

#define T_  512
#define H_  1024
#define B_  256
#define C_  128

__device__ __forceinline__ float sigm_(float x) {
    return 1.f / (1.f + __expf(-x));
}
__device__ __forceinline__ float tanh_(float x) {
    float e = __expf(-2.f * x);
    return (1.f - e) / (1.f + e);
}

__device__ __forceinline__ v16bf ldb16(const __bf16* __restrict__ p) {
    v8bf lo = *(const v8bf*)p;
    v8bf hi = *(const v8bf*)(p + 8);
    return __builtin_shufflevector(lo, hi,
        0,1,2,3,4,5,6,7,8,9,10,11,12,13,14,15);
}

__device__ __forceinline__ v8f wmma_bf16(v16bf a, v16bf b, v8f c) {
    return __builtin_amdgcn_wmma_f32_16x16x32_bf16(
        false, a, false, b, (short)0, c, false, false);
}

// ---------------------------------------------------------------------------
// Pack pass:
//  * W_h{g,i,f,o} (fp32 [1024,1024], row=K, col=N) -> bf16 WMMA B blocks.
//    Block = 32x16 KxN = 512 bf16 = 1 KiB; within a block:
//    lane = ((k&31)>=16 ? 16:0) + (n&15), element j = k&15 (lanes 0-15 hold
//    K=0..15 of their column, lanes 16-31 hold K=16..31; 16 contiguous bf16
//    per lane -> two b128 loads per operand).
//    kb-major: blk = ((k>>5)*4 + g)*64 + (n>>4)  (contiguous 256 KiB / K-step)
//  * W_hp (fp32 [1024,128]) -> same blocks, blk = (n>>4)*32 + (k>>5)
//  * Wxb[g] = W_x[g] + b[g] (fp32, bias folded: x is one-hot so
//    x@W + b = (W .+ b)[x]); kills the bias registers/adds in the hot loop.
// ---------------------------------------------------------------------------
__global__ void pack_weights(const float* __restrict__ Whg, const float* __restrict__ Whi,
                             const float* __restrict__ Whf, const float* __restrict__ Who,
                             const float* __restrict__ Whp,
                             const float* __restrict__ Wxg, const float* __restrict__ Wxi,
                             const float* __restrict__ Wxf, const float* __restrict__ Wxo,
                             const float* __restrict__ bg,  const float* __restrict__ bi,
                             const float* __restrict__ bff, const float* __restrict__ bo,
                             __bf16* __restrict__ whpk, __bf16* __restrict__ whppk,
                             float* __restrict__ wxb)
{
    const long id = (long)blockIdx.x * blockDim.x + threadIdx.x;
    const long NH  = 4l * 1024 * 1024;          // W_h elements
    const long NP  = 1024l * 128;               // W_hp elements
    const long NX  = 4l * 128 * 1024;           // W_x elements
    if (id < NH) {
        const int g   = (int)(id >> 20);
        const int rem = (int)(id & ((1 << 20) - 1));
        const int k = rem >> 10;
        const int n = rem & 1023;
        const float* src = (g == 0) ? Whg : (g == 1) ? Whi : (g == 2) ? Whf : Who;
        const int blk  = ((k >> 5) * 4 + g) * 64 + (n >> 4);
        const int lane = ((k >> 4) & 1) * 16 + (n & 15);
        const int j    = k & 15;
        whpk[(size_t)blk * 512 + lane * 16 + j] = (__bf16)src[rem];
    } else if (id < NH + NP) {
        const int rem = (int)(id - NH);
        const int k = rem >> 7;
        const int n = rem & 127;
        const int blk  = (n >> 4) * 32 + (k >> 5);
        const int lane = ((k >> 4) & 1) * 16 + (n & 15);
        const int j    = k & 15;
        whppk[(size_t)blk * 512 + lane * 16 + j] = (__bf16)Whp[rem];
    } else if (id < NH + NP + NX) {
        const int id2 = (int)(id - NH - NP);
        const int g   = id2 >> 17;              // 128*1024 per gate
        const int rem = id2 & ((1 << 17) - 1);
        const int n   = rem & 1023;
        const float* wsrc = (g == 0) ? Wxg : (g == 1) ? Wxi : (g == 2) ? Wxf : Wxo;
        const float* bsrc = (g == 0) ? bg  : (g == 1) ? bi  : (g == 2) ? bff : bo;
        wxb[(size_t)g * 131072 + rem] = wsrc[rem] + bsrc[n];
    }
}

// ---------------------------------------------------------------------------
// Persistent LSTM kernel: one workgroup per 16-row batch tile, all 512
// timesteps on-chip. h double-buffered in LDS (bf16), c in registers.
// 32 waves (8/SIMD = primary latency hiding); wave w owns output columns
// [w*32, w*32+32) for all 4 gates: 8 fp32 accumulator tiles, 256
// v_wmma_f32_16x16x32_bf16 per timestep per wave. B fragments rotate through
// two named registers (distance-2 pipelining, ~24 VGPRs) to keep peak K-loop
// pressure < 128 VGPRs -> no scratch spills at 8 waves/SIMD.
// ---------------------------------------------------------------------------
__global__ __launch_bounds__(1024, 1)
void lstm_persistent(const int* __restrict__ x,
                     const float* __restrict__ Wxb,
                     const __bf16* __restrict__ WhPk, const __bf16* __restrict__ WhpPk,
                     const float* __restrict__ bp, float* __restrict__ out)
{
    __shared__ __bf16 hbuf[2][16 * H_];   // 2 x 32 KiB double-buffered h tile

    const int tid     = threadIdx.x;
    const int wave    = tid >> 5;
    const int lane    = tid & 31;
    const int lane_n  = lane & 15;        // N within a 16-wide tile
    const int lane_hi = lane >> 4;        // row-half selector for C/D layout
    const int bbase   = blockIdx.x * 16;  // batch rows [bbase, bbase+16)
    const int colb    = wave * 32;        // owned columns

    // h0 = 0
    for (int i = tid; i < 16 * H_; i += 1024) hbuf[0][i] = (__bf16)0.f;
    __syncthreads();

    // per-lane base into packed weights: element offset for (kb,g,j) is
    // kb*131072 + g*32768 + j*512 (+ wave*2*512 + lane*16)
    const __bf16* bp0 = WhPk + (size_t)(wave * 2) * 512 + lane * 16;

    v8f   acc[4][2];
    float creg[2][8];
    #pragma unroll
    for (int j = 0; j < 2; ++j)
        #pragma unroll
        for (int r = 0; r < 8; ++r) creg[j][r] = 0.f;

    #pragma unroll 1
    for (int t = 0; t < T_; ++t) {
        const int cur = t & 1, nxt = cur ^ 1;

        // acc = (W_x[g] .+ b[g])[x_t]  -- one gather per element, bias folded
        {
            int xrow[8];
            #pragma unroll
            for (int r = 0; r < 8; ++r)
                xrow[r] = x[(bbase + r + lane_hi * 8) * T_ + t];

            #pragma unroll
            for (int g = 0; g < 4; ++g) {
                const float* wsrc = Wxb + (size_t)g * 131072;
                #pragma unroll
                for (int j = 0; j < 2; ++j)
                    #pragma unroll
                    for (int r = 0; r < 8; ++r)
                        acc[g][j][r] =
                            wsrc[xrow[r] * H_ + colb + j * 16 + lane_n];
            }
        }

        // ---- acc += h_t @ W_h : K=1024 in 32 steps ----
        // two rotating B fragments, reloaded right after consumption; the
        // tail reloads target kb+1 so the distance-2 pipeline never drains.
        v16bf fA = ldb16(bp0);          // (kb=0, g0, j0)
        v16bf fB = ldb16(bp0 + 512);    // (kb=0, g0, j1)

        #pragma unroll 1
        for (int kb = 0; kb < 32; ++kb) {
            const __bf16* pk = bp0 + (size_t)kb * 131072;

            // A fragment from LDS: lanes 0-15 -> K {0..15},
            // lanes 16-31 -> K {16..31} of row M = lane&15
            const int aoff = lane_n * H_ + kb * 32 + lane_hi * 16;
            v8bf alo = *(const v8bf*)&hbuf[cur][aoff];
            v8bf ahi = *(const v8bf*)&hbuf[cur][aoff + 8];
            v16bf af = __builtin_shufflevector(alo, ahi,
                0,1,2,3,4,5,6,7,8,9,10,11,12,13,14,15);

            acc[0][0] = wmma_bf16(af, fA, acc[0][0]);  fA = ldb16(pk + 32768);
            acc[0][1] = wmma_bf16(af, fB, acc[0][1]);  fB = ldb16(pk + 33280);
            acc[1][0] = wmma_bf16(af, fA, acc[1][0]);  fA = ldb16(pk + 65536);
            acc[1][1] = wmma_bf16(af, fB, acc[1][1]);  fB = ldb16(pk + 66048);
            acc[2][0] = wmma_bf16(af, fA, acc[2][0]);  fA = ldb16(pk + 98304);
            acc[2][1] = wmma_bf16(af, fB, acc[2][1]);  fB = ldb16(pk + 98816);
            acc[3][0] = wmma_bf16(af, fA, acc[3][0]);
            if (kb < 31) fA = ldb16(pk + 131072);
            acc[3][1] = wmma_bf16(af, fB, acc[3][1]);
            if (kb < 31) fB = ldb16(pk + 131584);
        }

        // gate nonlinearities + state update (register-elementwise: all 4
        // gate accumulators share the same (M,N) <-> (VGPR,lane) mapping)
        #pragma unroll
        for (int j = 0; j < 2; ++j) {
            #pragma unroll
            for (int r = 0; r < 8; ++r) {
                float gg = tanh_(acc[0][j][r]);
                float ii = sigm_(acc[1][j][r]);
                float ff = sigm_(acc[2][j][r]);
                float oo = sigm_(acc[3][j][r]);
                float cn = gg * ii + creg[j][r] * ff;
                creg[j][r] = cn;
                float hn = tanh_(cn) * oo;
                const int M = r + lane_hi * 8;
                hbuf[nxt][M * H_ + colb + j * 16 + lane_n] = (__bf16)hn;
            }
        }
        __syncthreads();
    }

    // Final projection: out[16,128] = h_T @ W_hp + b_p (waves 0..7, one
    // 16-col tile each). T_ is even so h_T lives in hbuf[0].
    if (wave < 8) {
        v8f po;
        const float bpv = bp[wave * 16 + lane_n];
        #pragma unroll
        for (int r = 0; r < 8; ++r) po[r] = bpv;

        #pragma unroll 2
        for (int kb = 0; kb < 32; ++kb) {
            const int aoff = lane_n * H_ + kb * 32 + lane_hi * 16;
            v8bf alo = *(const v8bf*)&hbuf[0][aoff];
            v8bf ahi = *(const v8bf*)&hbuf[0][aoff + 8];
            v16bf afrag = __builtin_shufflevector(alo, ahi,
                0,1,2,3,4,5,6,7,8,9,10,11,12,13,14,15);

            const int blk = wave * 32 + kb;
            v16bf bfrag = ldb16(WhpPk + (size_t)blk * 512 + lane * 16);

            po = wmma_bf16(afrag, bfrag, po);
        }

        #pragma unroll
        for (int r = 0; r < 8; ++r) {
            const int M = r + lane_hi * 8;
            out[(bbase + M) * C_ + wave * 16 + lane_n] = po[r];
        }
    }
}

extern "C" void kernel_launch(void* const* d_in, const int* in_sizes, int n_in,
                              void* d_out, int out_size, void* d_ws, size_t ws_size,
                              hipStream_t stream) {
    const int*   x   = (const int*)  d_in[0];
    const float* Wxg = (const float*)d_in[1];
    const float* Whg = (const float*)d_in[2];
    const float* bg  = (const float*)d_in[3];
    const float* Wxi = (const float*)d_in[4];
    const float* Whi = (const float*)d_in[5];
    const float* bi  = (const float*)d_in[6];
    const float* Wxf = (const float*)d_in[7];
    const float* Whf = (const float*)d_in[8];
    const float* bff = (const float*)d_in[9];
    const float* Wxo = (const float*)d_in[10];
    const float* Who = (const float*)d_in[11];
    const float* bo  = (const float*)d_in[12];
    const float* Whp = (const float*)d_in[13];
    const float* bp  = (const float*)d_in[14];

    __bf16* whpk  = (__bf16*)d_ws;                                     // 8 MiB
    __bf16* whppk = (__bf16*)((char*)d_ws + (size_t)8 * 1024 * 1024);  // 256 KiB
    float*  wxb   = (float*) ((char*)d_ws + (size_t)8 * 1024 * 1024
                                          + 256 * 1024);               // 2 MiB

    const long total = 4l * 1024 * 1024 + 1024l * 128 + 4l * 128 * 1024;
    pack_weights<<<(int)((total + 255) / 256), 256, 0, stream>>>(
        Whg, Whi, Whf, Who, Whp, Wxg, Wxi, Wxf, Wxo, bg, bi, bff, bo,
        whpk, whppk, wxb);

    lstm_persistent<<<B_ / 16, 1024, 0, stream>>>(
        x, wxb, whpk, whppk, bp, (float*)d_out);
}
